// SwinTransformerBlock4D_90967407329710
// MI455X (gfx1250) — matrix operand order repaired and checked by reading, not compiled, not run on hardware
//
#include <hip/hip_runtime.h>
#include <hip/hip_bf16.h>

// ---------------------------------------------------------------------------
// SwinTransformerBlock4D (Mamba mixer) for gfx1250 / MI455X.
// All dense GEMMs go through v_wmma_f32_16x16x32_f16 (wave32 WMMA),
// 16x32 output tile per wave (2 accumulators sharing one A fragment),
// software-pipelined fragment loads so WMMA overlaps global loads.
// ---------------------------------------------------------------------------

typedef __attribute__((ext_vector_type(16))) _Float16 v16h;
typedef __attribute__((ext_vector_type(8)))  float    v8f;
typedef __attribute__((ext_vector_type(4)))  _Float16 h4;

#define TOK   131072     // 2 * 16^4 tokens
#define DIMC  128
#define DI    256
#define NST   16
#define KC    4
#define DTR   8
#define MLPD  512

// ------------------------- WMMA fragment loaders ---------------------------
// A: [M,K] f16 row-major, 16x32 tile. Per ISA 7.12.2:
//   lane m = l&15; lanes 0-15: K pairs 0..7 and 16..23; lanes 16-31: 8..15, 24..31
__device__ __forceinline__ v16h load_frag_a(const _Float16* __restrict__ A,
                                            int K, int row0, int k0, int lane) {
  union { v16h v; unsigned u[8]; } f;
  int m = row0 + (lane & 15);
  const unsigned* base = (const unsigned*)(A + (size_t)m * K + k0);
  int hb = (lane & 16) ? 1 : 0;
  int d0 = hb ? 4 : 0;    // dword index of first K-pair group
  int d1 = hb ? 12 : 8;
#pragma unroll
  for (int p = 0; p < 4; ++p) f.u[p] = base[d0 + p];
#pragma unroll
  for (int p = 0; p < 4; ++p) f.u[4 + p] = base[d1 + p];
  return f.v;
}

// B from W[N,K] row-major (computing A @ W^T): B[k][n] = W[n][k].
// 32x16 f16 B tile: lane n = l&15; lanes 0-15: K 0..7 and 16..23; lanes 16-31: 8..15, 24..31.
__device__ __forceinline__ v16h load_frag_b(const _Float16* __restrict__ W,
                                            int K, int n0, int k0, int lane) {
  union { v16h v; unsigned u[8]; } f;
  int n = n0 + (lane & 15);
  const unsigned* base = (const unsigned*)(W + (size_t)n * K + k0);
  int d0 = (lane & 16) ? 4 : 0;
#pragma unroll
  for (int p = 0; p < 4; ++p) f.u[p] = base[d0 + p];
#pragma unroll
  for (int p = 0; p < 4; ++p) f.u[4 + p] = base[d0 + 8 + p];
  return f.v;
}

// ------------------------------ epilogues ----------------------------------
// EPI 0: store f32
// EPI 1: in_proj split: n<256 -> u (f32), n>=256 -> silu(z) (f16)   [N==512]
// EPI 2: +bias, exact GELU, store f16
// EPI 3: +bias +residual, store f32
template <int EPI>
__device__ __forceinline__ void epi_store(float v, int m, int n, int N,
                                          const float* __restrict__ bias,
                                          const float* __restrict__ resid,
                                          float* __restrict__ out_f32,
                                          _Float16* __restrict__ out_f16) {
  if (EPI == 0) {
    out_f32[(size_t)m * N + n] = v;
  } else if (EPI == 1) {
    if (n < 256) {
      out_f32[(size_t)m * 256 + n] = v;              // u (pre-conv)
    } else {
      float s = v / (1.f + __expf(-v));              // silu(z)
      out_f16[(size_t)m * 256 + (n - 256)] = (_Float16)s;
    }
  } else if (EPI == 2) {
    v += bias[n];
    float g = 0.5f * v * (1.f + erff(v * 0.70710678118654752f));
    out_f16[(size_t)m * N + n] = (_Float16)g;
  } else {
    v += bias[n] + resid[(size_t)m * N + n];
    out_f32[(size_t)m * N + n] = v;
  }
}

// ------------------------------ GEMM kernel --------------------------------
// C[M,N] = A[M,K] @ W[N,K]^T, f16 in, f32 accumulate via WMMA.
// One wave computes a 16x32 output tile (two 16x16 WMMA accumulators that
// share the A fragment). Fragment loads for k+32 are issued before the WMMAs
// on the k fragments so matrix ops overlap outstanding global loads.
template <int EPI>
__global__ __launch_bounds__(256) void gemm_wmma_kernel(
    const _Float16* __restrict__ A, const _Float16* __restrict__ W,
    const float* __restrict__ bias, const float* __restrict__ resid,
    float* __restrict__ out_f32, _Float16* __restrict__ out_f16,
    int M, int N, int K) {
  int wave = threadIdx.x >> 5;
  int lane = threadIdx.x & 31;
  int ntn = N >> 5;                                   // 32-wide n tiles
  long tile = (long)blockIdx.x * 8 + wave;
  int mt = (int)(tile / ntn);
  int nt = (int)(tile % ntn);
  int m0 = mt * 16;
  int n0 = nt * 32;

  v8f c0 = {0.f, 0.f, 0.f, 0.f, 0.f, 0.f, 0.f, 0.f};
  v8f c1 = {0.f, 0.f, 0.f, 0.f, 0.f, 0.f, 0.f, 0.f};

  // prologue: fragments for k0 = 0
  v16h a  = load_frag_a(A, K, m0, 0, lane);
  v16h b0 = load_frag_b(W, K, n0,      0, lane);
  v16h b1 = load_frag_b(W, K, n0 + 16, 0, lane);

  for (int k0 = 32; k0 < K; k0 += 32) {
    __builtin_prefetch(A + (size_t)(m0 + (lane & 15)) * K + k0 + 32, 0, 3);
    // issue next-k loads before consuming current fragments
    v16h an  = load_frag_a(A, K, m0, k0, lane);
    v16h bn0 = load_frag_b(W, K, n0,      k0, lane);
    v16h bn1 = load_frag_b(W, K, n0 + 16, k0, lane);
    c0 = __builtin_amdgcn_wmma_f32_16x16x32_f16(false, a, false, b0,
                                                (short)0, c0, false, false);
    c1 = __builtin_amdgcn_wmma_f32_16x16x32_f16(false, a, false, b1,
                                                (short)0, c1, false, false);
    a = an; b0 = bn0; b1 = bn1;
  }
  c0 = __builtin_amdgcn_wmma_f32_16x16x32_f16(false, a, false, b0,
                                              (short)0, c0, false, false);
  c1 = __builtin_amdgcn_wmma_f32_16x16x32_f16(false, a, false, b1,
                                              (short)0, c1, false, false);

  int nA = n0 + (lane & 15);
  int nB = nA + 16;
  int mbase = m0 + ((lane & 16) ? 8 : 0);
#pragma unroll
  for (int r = 0; r < 8; ++r) {
    int m = mbase + r;
    epi_store<EPI>(c0[r], m, nA, N, bias, resid, out_f32, out_f16);
    epi_store<EPI>(c1[r], m, nB, N, bias, resid, out_f32, out_f16);
  }
}

// --------------------------- LayerNorm helper ------------------------------
__device__ __forceinline__ void ln128_store_f16(float4 v, const float* __restrict__ g,
                                                const float* __restrict__ b, int lane,
                                                _Float16* __restrict__ dst) {
  float s = v.x + v.y + v.z + v.w;
#pragma unroll
  for (int off = 16; off > 0; off >>= 1) s += __shfl_xor(s, off, 32);
  float mu = s * (1.f / 128.f);
  float d0 = v.x - mu, d1 = v.y - mu, d2 = v.z - mu, d3 = v.w - mu;
  float q = d0 * d0 + d1 * d1 + d2 * d2 + d3 * d3;
#pragma unroll
  for (int off = 16; off > 0; off >>= 1) q += __shfl_xor(q, off, 32);
  float inv = rsqrtf(q * (1.f / 128.f) + 1e-5f);
  int ch = lane * 4;
  h4 o;
  o.x = (_Float16)(d0 * inv * g[ch + 0] + b[ch + 0]);
  o.y = (_Float16)(d1 * inv * g[ch + 1] + b[ch + 1]);
  o.z = (_Float16)(d2 * inv * g[ch + 2] + b[ch + 2]);
  o.w = (_Float16)(d3 * inv * g[ch + 3] + b[ch + 3]);
  *((h4*)dst + lane) = o;
}

// -------------------- K1: LN1 + roll(-2) + window partition ----------------
__global__ __launch_bounds__(256) void ln_shift_part_kernel(
    const float* __restrict__ x, const float* __restrict__ g,
    const float* __restrict__ b, _Float16* __restrict__ wf16) {
  int wave = threadIdx.x >> 5, lane = threadIdx.x & 31;
  int t = blockIdx.x * 8 + wave;            // window-ordered token
  int l = t & 255, w = t >> 8;
  int bb = w >> 8;
  int wD = (w >> 6) & 3, wH = (w >> 4) & 3, wW = (w >> 2) & 3, wT = w & 3;
  int i0 = (l >> 6) & 3, i1 = (l >> 4) & 3, i2 = (l >> 2) & 3, i3 = l & 3;
  int D = (wD * 4 + i0 + 2) & 15;
  int H = (wH * 4 + i1 + 2) & 15;
  int Wd = (wW * 4 + i2 + 2) & 15;
  int T = (wT * 4 + i3 + 2) & 15;
  size_t src = ((((size_t)bb * 16 + D) * 16 + H) * 16 + Wd) * 16 + T;
  float4 v = *((const float4*)(x + src * 128) + lane);
  ln128_store_f16(v, g, b, lane, wf16 + (size_t)t * 128);
}

// -------------------- K3: causal depthwise conv + SiLU ---------------------
__global__ __launch_bounds__(256) void conv_silu_kernel(
    const float* __restrict__ u_raw, const float* __restrict__ cw,
    const float* __restrict__ cb, float* __restrict__ u_post) {
  int d = threadIdx.x;          // channel
  int t = blockIdx.x;           // window-ordered token
  int l = t & 255, wbase = t & ~255;
  float acc = cb[d];
#pragma unroll
  for (int j = 0; j < KC; ++j) {
    int ll = l - (KC - 1) + j;
    if (ll >= 0) acc += cw[d * KC + j] * u_raw[(size_t)(wbase + ll) * DI + d];
  }
  float s = acc / (1.f + __expf(-acc));
  u_post[(size_t)t * DI + d] = s;
}

// -------------------- K4: x_proj + dt_proj + softplus ----------------------
__global__ __launch_bounds__(256) void xproj_kernel(
    const float* __restrict__ u_post, const float* __restrict__ xw,
    const float* __restrict__ dtw, const float* __restrict__ dtb,
    float* __restrict__ delta, float* __restrict__ bc) {
  __shared__ float xd[DTR + 2 * NST];        // 40
  int t = blockIdx.x, tid = threadIdx.x;
  if (tid < DTR + 2 * NST) {
    const float* up = u_post + (size_t)t * DI;
    const float* wr = xw + tid * DI;
    float acc = 0.f;
    for (int k = 0; k < DI; ++k) acc += up[k] * wr[k];
    xd[tid] = acc;
  }
  __syncthreads();
  if (tid < 2 * NST) bc[(size_t)t * 32 + tid] = xd[DTR + tid];
  float acc = dtb[tid];
#pragma unroll
  for (int r = 0; r < DTR; ++r) acc += xd[r] * dtw[tid * DTR + r];
  float sp = (acc > 20.f) ? acc : log1pf(__expf(acc));
  delta[(size_t)t * DI + tid] = sp;
}

// -------------------- K5: selective scan (block per window) ----------------
__global__ __launch_bounds__(256) void scan_kernel(
    const float* __restrict__ delta, const float* __restrict__ bc,
    const float* __restrict__ u_post, const _Float16* __restrict__ sz,
    const float* __restrict__ A_log, const float* __restrict__ Dsk,
    _Float16* __restrict__ y_out) {
  __shared__ float BCs[256 * 32];            // 32 KB of the 320 KB WGP LDS
  int w = blockIdx.x, d = threadIdx.x;
  const float* bcw = bc + (size_t)w * 256 * 32;
  for (int i = d; i < 256 * 32; i += 256) BCs[i] = bcw[i];
  __syncthreads();
  float A[NST], h[NST];
#pragma unroll
  for (int n = 0; n < NST; ++n) { A[n] = -__expf(A_log[d * NST + n]); h[n] = 0.f; }
  float Dd = Dsk[d];
  size_t base = (size_t)w * 256;
  for (int t = 0; t < 256; ++t) {
    size_t idx = (base + t) * DI + d;
    float dt = delta[idx];
    float ut = u_post[idx];
    float du = dt * ut;
    const float* Bt = &BCs[t * 32];
    const float* Ct = Bt + NST;
    float y = 0.f;
#pragma unroll
    for (int n = 0; n < NST; ++n) {
      float dA = __expf(dt * A[n]);
      h[n] = dA * h[n] + du * Bt[n];
      y += h[n] * Ct[n];
    }
    float yf = (y + ut * Dd) * (float)sz[idx];
    y_out[idx] = (_Float16)yf;
  }
}

// ------- K7: window reverse + roll(+2) + residual + LN2 --------------------
__global__ __launch_bounds__(256) void rev_res_ln_kernel(
    const float* __restrict__ xin, const float* __restrict__ x1,
    const float* __restrict__ g, const float* __restrict__ b,
    float* __restrict__ xres, _Float16* __restrict__ xm16) {
  int wave = threadIdx.x >> 5, lane = threadIdx.x & 31;
  int tn = blockIdx.x * 8 + wave;           // natural-order token
  int T = tn & 15, Wd = (tn >> 4) & 15, H = (tn >> 8) & 15;
  int D = (tn >> 12) & 15, bb = tn >> 16;
  int pD = (D - 2) & 15, pH = (H - 2) & 15, pW = (Wd - 2) & 15, pT = (T - 2) & 15;
  int w = bb * 256 + (((pD >> 2) * 4 + (pH >> 2)) * 4 + (pW >> 2)) * 4 + (pT >> 2);
  int l = (((pD & 3) * 4 + (pH & 3)) * 4 + (pW & 3)) * 4 + (pT & 3);
  size_t swin = (size_t)(w * 256 + l);
  float4 a = *((const float4*)(xin + (size_t)tn * 128) + lane);
  float4 c = *((const float4*)(x1 + swin * 128) + lane);
  float4 v = {a.x + c.x, a.y + c.y, a.z + c.z, a.w + c.w};
  *((float4*)(xres + (size_t)tn * 128) + lane) = v;
  ln128_store_f16(v, g, b, lane, xm16 + (size_t)tn * 128);
}

// -------------------- weight f32 -> f16 conversion -------------------------
__global__ void cvt_f16_kernel(const float* __restrict__ a,
                               _Float16* __restrict__ o, int n) {
  int i = blockIdx.x * 256 + threadIdx.x;
  if (i < n) o[i] = (_Float16)a[i];
}

// ---------------------------------------------------------------------------
extern "C" void kernel_launch(void* const* d_in, const int* in_sizes, int n_in,
                              void* d_out, int out_size, void* d_ws, size_t ws_size,
                              hipStream_t stream) {
  const float* x        = (const float*)d_in[0];
  const float* n1g      = (const float*)d_in[1];
  const float* n1b      = (const float*)d_in[2];
  const float* in_proj  = (const float*)d_in[3];   // [512,128]
  const float* conv_w   = (const float*)d_in[4];   // [256,1,4]
  const float* conv_b   = (const float*)d_in[5];
  const float* x_proj   = (const float*)d_in[6];   // [40,256]
  const float* dt_projw = (const float*)d_in[7];   // [256,8]
  const float* dt_projb = (const float*)d_in[8];
  const float* A_log    = (const float*)d_in[9];   // [256,16]
  const float* D_skip   = (const float*)d_in[10];
  const float* out_proj = (const float*)d_in[11];  // [128,256]
  const float* n2g      = (const float*)d_in[12];
  const float* n2b      = (const float*)d_in[13];
  const float* fc1_w    = (const float*)d_in[14];  // [512,128]
  const float* fc1_b    = (const float*)d_in[15];
  const float* fc2_w    = (const float*)d_in[16];  // [128,512]
  const float* fc2_b    = (const float*)d_in[17];
  float* out            = (float*)d_out;

  // ---- workspace layout ----
  char* ws = (char*)d_ws;
  size_t off = 0;
  auto alloc = [&](size_t bytes) -> void* {
    void* p = ws + off;
    off = (off + bytes + 255) & ~(size_t)255;
    return p;
  };
  const size_t T = (size_t)TOK;
  _Float16* wf16  = (_Float16*)alloc(T * 128 * 2);   // LN1 tokens, f16
  _Float16* w_ip  = (_Float16*)alloc(512 * 128 * 2); // in_proj f16
  _Float16* w_op  = (_Float16*)alloc(128 * 256 * 2); // out_proj f16
  _Float16* w_f1  = (_Float16*)alloc(512 * 128 * 2); // fc1 f16
  _Float16* w_f2  = (_Float16*)alloc(128 * 512 * 2); // fc2 f16
  float*    u_raw = (float*)   alloc(T * 256 * 4);   // pre-conv u
  _Float16* szb   = (_Float16*)alloc(T * 256 * 2);   // silu(z)
  float*    u_pst = (float*)   alloc(T * 256 * 4);   // post conv+silu
  float*    dlt   = (float*)   alloc(T * 256 * 4);   // softplus delta
  float*    bcb   = (float*)   alloc(T * 32 * 4);    // B,C per token
  _Float16* y16   = (_Float16*)alloc(T * 256 * 2);   // gated scan output
  float*    x1b   = (float*)   alloc(T * 128 * 4);   // out_proj result (window order)
  float*    xres  = (float*)   alloc(T * 128 * 4);   // residual after part1 (natural)
  _Float16* xm16  = (_Float16*)alloc(T * 128 * 2);   // LN2 f16
  _Float16* h16   = (_Float16*)alloc(T * 512 * 2);   // MLP hidden f16
  (void)ws_size; (void)n_in; (void)in_sizes; (void)out_size;

  // ---- weight conversion ----
  cvt_f16_kernel<<<(512 * 128 + 255) / 256, 256, 0, stream>>>(in_proj, w_ip, 512 * 128);
  cvt_f16_kernel<<<(128 * 256 + 255) / 256, 256, 0, stream>>>(out_proj, w_op, 128 * 256);
  cvt_f16_kernel<<<(512 * 128 + 255) / 256, 256, 0, stream>>>(fc1_w, w_f1, 512 * 128);
  cvt_f16_kernel<<<(128 * 512 + 255) / 256, 256, 0, stream>>>(fc2_w, w_f2, 128 * 512);

  // ---- K1: LN1 + shift + window partition (wave per token) ----
  ln_shift_part_kernel<<<TOK / 8, 256, 0, stream>>>(x, n1g, n1b, wf16);

  // ---- K2: in_proj WMMA GEMM, split u / silu(z) ----
  // tiles(16x32) = (TOK/16)*(512/32) = 131072; 8 waves/block -> 16384 blocks
  gemm_wmma_kernel<1><<<(TOK / 16) * (512 / 32) / 8, 256, 0, stream>>>(
      wf16, w_ip, nullptr, nullptr, u_raw, szb, TOK, 512, 128);

  // ---- K3: depthwise causal conv + SiLU ----
  conv_silu_kernel<<<TOK, 256, 0, stream>>>(u_raw, conv_w, conv_b, u_pst);

  // ---- K4: x_proj + dt_proj + softplus ----
  xproj_kernel<<<TOK, 256, 0, stream>>>(u_pst, x_proj, dt_projw, dt_projb, dlt, bcb);

  // ---- K5: selective scan, fused D-skip + silu(z) gate ----
  scan_kernel<<<512, 256, 0, stream>>>(dlt, bcb, u_pst, szb, A_log, D_skip, y16);

  // ---- K6: out_proj WMMA GEMM ----
  gemm_wmma_kernel<0><<<(TOK / 16) * (128 / 32) / 8, 256, 0, stream>>>(
      y16, w_op, nullptr, nullptr, x1b, nullptr, TOK, 128, 256);

  // ---- K7: window reverse + unshift + residual + LN2 ----
  rev_res_ln_kernel<<<TOK / 8, 256, 0, stream>>>(x, x1b, n2g, n2b, xres, xm16);

  // ---- K8: fc1 WMMA GEMM + exact GELU ----
  gemm_wmma_kernel<2><<<(TOK / 16) * (512 / 32) / 8, 256, 0, stream>>>(
      xm16, w_f1, fc1_b, nullptr, nullptr, h16, TOK, 512, 128);

  // ---- K9: fc2 WMMA GEMM + bias + residual -> d_out ----
  gemm_wmma_kernel<3><<<(TOK / 16) * (128 / 32) / 8, 256, 0, stream>>>(
      h16, w_f2, fc2_b, xres, out, nullptr, TOK, 128, 512);
}
